// ARONet_21869973471686
// MI455X (gfx1250) — matrix operand order repaired and checked by reading, not compiled, not run on hardware
//
#include <hip/hip_runtime.h>
#include <math.h>

// ---------------- problem constants ----------------
#define N_BS   2
#define N_PTS  2048
#define N_QRY  128
#define N_ANC  48
#define N_LOC  32
#define HID    128
#define DM     128
#define DFF    2048
#define NL     6
#define NH     8
#define HD     16
#define SAMPLES (N_BS*N_QRY*N_ANC)   /* 12288 */
#define R_TOT   (SAMPLES*N_LOC)      /* 393216 */
#define NCHUNK  8
#define SCHUNK  (SAMPLES/NCHUNK)     /* 1536 samples */
#define RCHUNK  (SCHUNK*N_LOC)       /* 49152 rows */
#define COS_TH  0.86602540378443864676f   /* cos(pi/6) */
#define KCAT    192                       /* fc1-conv K padded 132 -> 192 (mult of 64) */

// GEMM blocking: 8 waves/block, each wave 32(M)x64(N), block = 256(M)x64(N), 64-K panels
#define GBM 256
#define GBN 64

// ---------------- CDNA5 WMMA types ----------------
typedef __attribute__((ext_vector_type(16))) __bf16 v16bf;
typedef __attribute__((ext_vector_type(8)))  __bf16 v8bf;
typedef __attribute__((ext_vector_type(8)))  float  v8f;

union BF16x16u { v16bf v; v8bf h[2]; unsigned short s[16]; };

__device__ __forceinline__ __bf16 f2bf(float f) {
  union { float f; unsigned u; } x; x.f = f;
  unsigned r = x.u + 0x7FFFu + ((x.u >> 16) & 1u);   // RNE
  union { unsigned short s; __bf16 b; } y; y.s = (unsigned short)(r >> 16);
  return y.b;
}

// ---------------- bf16 WMMA GEMM with async-LDS weight staging ----------------
// D[M,N] = (RELU? relu(A) : A)[M,K] * W[N,K]^T + bias[N] (+ Cin[M,N])
//   optional f32 out (Cout) and fused relu->bf16 out (Obf).
// Requires M%256==0, N%64==0, K%64==0 (all call sites satisfy this).
// B panel (64 cols x 64 K, 8KB) double-buffered in LDS via
// GLOBAL_LOAD_ASYNC_TO_LDS_B128 (ASYNCcnt) + workgroup split-barrier:
// 16 WMMAs per sync. Fragments follow the ISA 16-bit A(16x32)/B(32x16)
// wave32 VGPR layouts; every ds_load is consumed by a WMMA before the
// end-of-iteration barrier, so buffer reuse is hazard-free.
template<bool RELU>
__global__ void __launch_bounds__(256)
k_gemm_bf16(const __bf16* __restrict__ A,
            const __bf16* __restrict__ W,
            const float*  __restrict__ bias,
            const float*  __restrict__ Cin,
            float*        __restrict__ Cout,
            __bf16*       __restrict__ Obf,
            int M, int K, int N)
{
  __shared__ __align__(16) unsigned char smem[2 * 8192];

  const int wave = threadIdx.x >> 5;
  const int lane = threadIdx.x & 31;
  const int nb   = N / GBN;
  const int bm   = blockIdx.x / nb;
  const int bn   = blockIdx.x % nb;
  const int nbase = bn * GBN;
  const int rbase = bm * GBM + wave * 32;

  // --- cooperative async B-panel stage: 8KB = 512 b128 chunks, 2 per thread
  const int c0 = threadIdx.x;          // chunk ids
  const int c1 = threadIdx.x + 256;
  const __bf16* gB0 = W + (size_t)(nbase + (c0 >> 3)) * K + (c0 & 7) * 8;
  const __bf16* gB1 = W + (size_t)(nbase + (c1 >> 3)) * K + (c1 & 7) * 8;
  const unsigned l0 = (unsigned)(c0 * 16);
  const unsigned l1 = (unsigned)(c1 * 16);

  // --- A fragment pointers (two 16-row tiles per wave)
  const int aksel = (lane >> 4) * 8;
  const __bf16* A0 = A + (size_t)(rbase + (lane & 15)) * K;
  const __bf16* A1 = A + (size_t)(rbase + 16 + (lane & 15)) * K;

  // --- B fragment LDS addressing: col stride 128B; lane group selects K half
  const int bcol = lane & 15;
  const int bks  = (lane >> 4) * 16;        // bytes

  v8f acc[2][4] = {};

  auto stage = [&](int k0, int buf) {
    unsigned d0 = (unsigned)(buf * 8192) + l0;
    unsigned d1 = (unsigned)(buf * 8192) + l1;
    const __bf16* s0 = gB0 + k0;
    const __bf16* s1 = gB1 + k0;
    asm volatile("global_load_async_to_lds_b128 %0, %2, off\n\t"
                 "global_load_async_to_lds_b128 %1, %3, off"
                 :: "v"(d0), "v"(d1), "v"(s0), "v"(s1) : "memory");
  };

  stage(0, 0);
  asm volatile("s_wait_asynccnt 0x0" ::: "memory");
  __syncthreads();

  int buf = 0;
  for (int k0 = 0; k0 < K; k0 += 64, buf ^= 1) {
    if (k0 + 64 < K) stage(k0 + 64, buf ^ 1);

    const unsigned char* bufp = &smem[buf * 8192];
#pragma unroll
    for (int kh = 0; kh < 2; ++kh) {
      const int ka = k0 + kh * 32;
      BF16x16u a0, a1;
      a0.h[0] = *reinterpret_cast<const v8bf*>(A0 + ka + aksel);
      a0.h[1] = *reinterpret_cast<const v8bf*>(A0 + ka + 16 + aksel);
      a1.h[0] = *reinterpret_cast<const v8bf*>(A1 + ka + aksel);
      a1.h[1] = *reinterpret_cast<const v8bf*>(A1 + ka + 16 + aksel);
      if (RELU) {
#pragma unroll
        for (int i = 0; i < 16; ++i) {
          a0.s[i] = (a0.s[i] & 0x8000u) ? (unsigned short)0 : a0.s[i];
          a1.s[i] = (a1.s[i] & 0x8000u) ? (unsigned short)0 : a1.s[i];
        }
      }

      // all four B tiles live simultaneously (distinct SRC1 register sets)
      BF16x16u b[4];
#pragma unroll
      for (int tn = 0; tn < 4; ++tn) {
        const unsigned char* bp = bufp + (tn * 16 + bcol) * 128 + kh * 64 + bks;
        b[tn].h[0] = *reinterpret_cast<const v8bf*>(bp);        // ds_load_b128
        b[tn].h[1] = *reinterpret_cast<const v8bf*>(bp + 32);   // ds_load_b128
      }
#pragma unroll
      for (int tn = 0; tn < 4; ++tn) {
        acc[0][tn] = __builtin_amdgcn_wmma_f32_16x16x32_bf16(
            false, a0.v, false, b[tn].v, (short)0, acc[0][tn], false, false);
        acc[1][tn] = __builtin_amdgcn_wmma_f32_16x16x32_bf16(
            false, a1.v, false, b[tn].v, (short)0, acc[1][tn], false, false);
      }
    }

    asm volatile("s_wait_asynccnt 0x0" ::: "memory");
    __syncthreads();
  }

  // --- epilogue (C/D layout: lane 0-15 -> M 0..7, lane 16-31 -> M 8..15)
  const int rowoff = (lane >> 4) * 8;
#pragma unroll
  for (int mt = 0; mt < 2; ++mt) {
#pragma unroll
    for (int tn = 0; tn < 4; ++tn) {
      const int col = nbase + tn * 16 + (lane & 15);
      const float bv = bias ? bias[col] : 0.0f;
      const int r0 = rbase + mt * 16 + rowoff;
#pragma unroll
      for (int i = 0; i < 8; ++i) {
        size_t idx = (size_t)(r0 + i) * N + col;
        float v = acc[mt][tn][i] + bv;
        if (Cin)  v += Cin[idx];
        if (Cout) Cout[idx] = v;
        if (Obf)  Obf[idx] = f2bf(fmaxf(v, 0.0f));
      }
    }
  }
}

// ---------------- weight conversion ----------------
__global__ void k_cvt_bf16(const float* __restrict__ s, __bf16* __restrict__ d, long n) {
  long i = (long)blockIdx.x * blockDim.x + threadIdx.x;
  if (i < n) d[i] = f2bf(s[i]);
}
__global__ void k_cvt_pad(const float* __restrict__ s, __bf16* __restrict__ d,
                          int N, int Kin, int Kout) {
  int i = blockIdx.x * blockDim.x + threadIdx.x;
  if (i >= N * Kout) return;
  int k = i % Kout, n = i / Kout;
  d[i] = f2bf(k < Kin ? s[n * Kin + k] : 0.0f);
}

// ---------------- cone cast + top-32 + relative feats ----------------
__global__ void k_cone_topk(const float* __restrict__ pcd,  // [b,2048,3]
                            const float* __restrict__ qry,  // [b,128,3]
                            const float* __restrict__ anc,  // [b,48,3]
                            float* __restrict__ feat_qh,    // [b,q,a,32,4]
                            float* __restrict__ feat_aq)    // [b,q,a,4]
{
  __shared__ float dist[N_PTS];
  __shared__ float rval[256];
  __shared__ int   ridx[256];
  const int id = blockIdx.x;                 // (b,q,a) flat, a fastest
  const int a = id % N_ANC;
  const int q = (id / N_ANC) % N_QRY;
  const int b = id / (N_ANC * N_QRY);
  const int tid = threadIdx.x;

  const float ax = anc[(b*N_ANC + a)*3 + 0];
  const float ay = anc[(b*N_ANC + a)*3 + 1];
  const float az = anc[(b*N_ANC + a)*3 + 2];
  const float qx = qry[(b*N_QRY + q)*3 + 0];
  const float qy = qry[(b*N_QRY + q)*3 + 1];
  const float qz = qry[(b*N_QRY + q)*3 + 2];
  const float vx = qx - ax, vy = qy - ay, vz = qz - az;
  const float vm = sqrtf(vx*vx + vy*vy + vz*vz);
  const float nx = vx / vm, ny = vy / vm, nz = vz / vm;

  const float* P = pcd + (size_t)b * N_PTS * 3;
  for (int p = tid; p < N_PTS; p += 256) {
    float px = P[p*3+0] - ax, py = P[p*3+1] - ay, pz = P[p*3+2] - az;
    float mod = sqrtf(px*px + py*py + pz*pz);
    float cosv = (px*nx + py*ny + pz*nz) / mod;
    dist[p] = (cosv <= COS_TH) ? __builtin_inff() : mod;
  }
  if (tid == 0) {
    size_t fa = ((size_t)(b*N_QRY + q)*N_ANC + a) * 4;
    feat_aq[fa+0] = nx; feat_aq[fa+1] = ny; feat_aq[fa+2] = nz; feat_aq[fa+3] = vm;
  }
  __syncthreads();

  const size_t obase = (((size_t)(b*N_QRY + q)*N_ANC + a) * N_LOC) * 4;
  for (int sel = 0; sel < N_LOC; ++sel) {
    float best = __builtin_inff(); int bidx = 0x7fffffff;
    for (int p = tid; p < N_PTS; p += 256) {
      float d = dist[p];
      if (d < best) { best = d; bidx = p; }
    }
    rval[tid] = best; ridx[tid] = bidx;
    __syncthreads();
    for (int off = 128; off > 0; off >>= 1) {
      if (tid < off) {
        if (rval[tid+off] < rval[tid]) { rval[tid] = rval[tid+off]; ridx[tid] = ridx[tid+off]; }
      }
      __syncthreads();
    }
    if (tid == 0) {
      float d = rval[0]; int pi = ridx[0];
      float hx, hy, hz;
      if (__builtin_isinf(d) || pi >= N_PTS) { hx = qx; hy = qy; hz = qz; }
      else {
        hx = P[pi*3+0]; hy = P[pi*3+1]; hz = P[pi*3+2];
        dist[pi] = __builtin_inff();
      }
      float dx = hx - qx, dy = hy - qy, dz = hz - qz;
      float sq = dx*dx + dy*dy + dz*dz;
      float m = (sq == 0.0f) ? 0.0f : sqrtf(sq);
      float safe = (sq == 0.0f) ? 1.0f : m;
      feat_qh[obase + sel*4 + 0] = dx / safe;
      feat_qh[obase + sel*4 + 1] = dy / safe;
      feat_qh[obase + sel*4 + 2] = dz / safe;
      feat_qh[obase + sel*4 + 3] = m;
    }
    __syncthreads();
  }
}

// ---------------- pointnet helpers ----------------
// fc_pos: [rows,4] @ W[256,4]^T + b -> bf16 [rows,256]   (K=4, VALU)
__global__ void k_fc_pos(const float* __restrict__ feat, const float* __restrict__ W,
                         const float* __restrict__ b, __bf16* __restrict__ out, int rows)
{
  __shared__ float f[4];
  int row = blockIdx.x;
  if (row >= rows) return;
  if (threadIdx.x < 4) f[threadIdx.x] = feat[(size_t)row*4 + threadIdx.x];
  __syncthreads();
  int o = threadIdx.x;  // 256
  float acc = b[o] + f[0]*W[o*4+0] + f[1]*W[o*4+1] + f[2]*W[o*4+2] + f[3]*W[o*4+3];
  out[(size_t)row*256 + o] = f2bf(acc);
}

// between blocks: x_next = [net | broadcast(max over 32 local pts)]
__global__ void k_pool_concat(const float* __restrict__ net, __bf16* __restrict__ out) {
  int s = blockIdx.x, c = threadIdx.x;   // 128 threads
  float mx = -1e30f;
  for (int j = 0; j < N_LOC; ++j) {
    float v = net[((size_t)s*N_LOC + j)*HID + c];
    mx = fmaxf(mx, v);
    out[((size_t)s*N_LOC + j)*(2*HID) + c] = f2bf(v);
  }
  __bf16 mb = f2bf(mx);
  for (int j = 0; j < N_LOC; ++j)
    out[((size_t)s*N_LOC + j)*(2*HID) + HID + c] = mb;
}

// final: relu(max over 32) -> bf16 [S,128]
__global__ void k_pool_relu(const float* __restrict__ net, __bf16* __restrict__ out) {
  int s = blockIdx.x, c = threadIdx.x;
  float mx = -1e30f;
  for (int j = 0; j < N_LOC; ++j)
    mx = fmaxf(mx, net[((size_t)s*N_LOC + j)*HID + c]);
  out[(size_t)s*HID + c] = f2bf(fmaxf(mx, 0.0f));
}

// concat [feat_aq | fl] in (b,a,q) row order, zero-pad K 132->KCAT
__global__ void k_concat_fc1(const float* __restrict__ feat_aq,  // [(b,q,a),4]
                             const float* __restrict__ fl,       // [(b,q,a),128]
                             __bf16* __restrict__ out)           // [(b,a,q),KCAT]
{
  int r = blockIdx.x;                        // (b,a,q)
  int q = r % N_QRY, a = (r / N_QRY) % N_ANC, b = r / (N_QRY * N_ANC);
  int rin = (b*N_QRY + q)*N_ANC + a;
  int c = threadIdx.x;                       // KCAT threads
  float v;
  if (c < 4)        v = feat_aq[(size_t)rin*4 + c];
  else if (c < 132) v = fl[(size_t)rin*HID + (c - 4)];
  else              v = 0.0f;
  out[(size_t)r*KCAT + c] = f2bf(v);
}

// eval-mode BN + ReLU -> bf16, same row order
__global__ void k_bn_relu(const float* __restrict__ y, const float* __restrict__ g,
                          const float* __restrict__ be, __bf16* __restrict__ out, long n)
{
  long i = (long)blockIdx.x * blockDim.x + threadIdx.x;
  if (i >= n) return;
  int c = (int)(i % HID);
  float v = y[i] * (g[c] * rsqrtf(1.0f + 1e-5f)) + be[c];
  out[i] = f2bf(fmaxf(v, 0.0f));
}

// BN + ReLU + transpose (b,a,q)->(b,q,a), write f32 and bf16
__global__ void k_bn_relu_tr(const float* __restrict__ y, const float* __restrict__ g,
                             const float* __restrict__ be,
                             float* __restrict__ xf, __bf16* __restrict__ xb)
{
  int rin = blockIdx.x;                      // (b,a,q)
  int q = rin % N_QRY, a = (rin / N_QRY) % N_ANC, b = rin / (N_QRY * N_ANC);
  int rout = (b*N_QRY + q)*N_ANC + a;
  int c = threadIdx.x;                       // 128
  float v = y[(size_t)rin*HID + c] * (g[c] * rsqrtf(1.0f + 1e-5f)) + be[c];
  v = fmaxf(v, 0.0f);
  xf[(size_t)rout*HID + c] = v;
  xb[(size_t)rout*HID + c] = f2bf(v);
}

// ---------------- transformer helpers ----------------
// attention: one block per (sample, head); S=48, HD=16
__global__ void k_attn(const float* __restrict__ qkv,   // [(b,q),48,384]
                       __bf16* __restrict__ out)        // [(b,q)*48,128]
{
  __shared__ float Ks[N_ANC][HD];
  __shared__ float Vs[N_ANC][HD];
  const int smp = blockIdx.x >> 3;
  const int h   = blockIdx.x & 7;
  const float* base = qkv + (size_t)smp * N_ANC * (3*DM);
  const int tid = threadIdx.x;               // 64 threads
  for (int t = tid; t < N_ANC*HD; t += 64) {
    int j = t >> 4, d = t & 15;
    Ks[j][d] = base[j*(3*DM) + DM   + h*HD + d];
    Vs[j][d] = base[j*(3*DM) + 2*DM + h*HD + d];
  }
  __syncthreads();
  if (tid < N_ANC) {
    float qv[HD];
#pragma unroll
    for (int d = 0; d < HD; ++d) qv[d] = base[tid*(3*DM) + h*HD + d];
    float sc[N_ANC];
    float mx = -1e30f;
    for (int j = 0; j < N_ANC; ++j) {
      float s = 0.f;
#pragma unroll
      for (int d = 0; d < HD; ++d) s += qv[d] * Ks[j][d];
      s *= 0.25f;                            // 1/sqrt(16)
      sc[j] = s; mx = fmaxf(mx, s);
    }
    float sum = 0.f;
    for (int j = 0; j < N_ANC; ++j) { sc[j] = __expf(sc[j] - mx); sum += sc[j]; }
    float inv = 1.0f / sum;
    float o[HD];
#pragma unroll
    for (int d = 0; d < HD; ++d) o[d] = 0.f;
    for (int j = 0; j < N_ANC; ++j)
#pragma unroll
      for (int d = 0; d < HD; ++d) o[d] += sc[j] * Vs[j][d];
#pragma unroll
    for (int d = 0; d < HD; ++d)
      out[((size_t)smp*N_ANC + tid)*DM + h*HD + d] = f2bf(o[d] * inv);
  }
}

// residual add + LayerNorm -> f32 and bf16
__global__ void k_add_ln(const float* __restrict__ x, const float* __restrict__ hh,
                         const float* __restrict__ g, const float* __restrict__ b,
                         float* __restrict__ of, __bf16* __restrict__ ob)
{
  __shared__ float red[DM];
  __shared__ float smean, svar;
  int r = blockIdx.x, c = threadIdx.x;       // 128 threads
  size_t idx = (size_t)r*DM + c;
  float v = x[idx] + hh[idx];
  red[c] = v; __syncthreads();
  for (int o = 64; o > 0; o >>= 1) { if (c < o) red[c] += red[c+o]; __syncthreads(); }
  if (c == 0) smean = red[0] * (1.0f / DM);
  __syncthreads();
  float d = v - smean;
  red[c] = d * d; __syncthreads();
  for (int o = 64; o > 0; o >>= 1) { if (c < o) red[c] += red[c+o]; __syncthreads(); }
  if (c == 0) svar = red[0] * (1.0f / DM);
  __syncthreads();
  float o = d * rsqrtf(svar + 1e-5f) * g[c] + b[c];
  of[idx] = o; ob[idx] = f2bf(o);
}

// occ[b,q] = dot(x[(b,q), 48*128], w) + bias
__global__ void k_out(const float* __restrict__ x, const float* __restrict__ w,
                      const float* __restrict__ bias, float* __restrict__ occ)
{
  __shared__ float red[256];
  int r = blockIdx.x;
  float acc = 0.f;
  for (int t = threadIdx.x; t < N_ANC*DM; t += 256)
    acc += x[(size_t)r*(N_ANC*DM) + t] * w[t];
  red[threadIdx.x] = acc; __syncthreads();
  for (int o = 128; o > 0; o >>= 1) {
    if (threadIdx.x < o) red[threadIdx.x] += red[threadIdx.x+o];
    __syncthreads();
  }
  if (threadIdx.x == 0) occ[r] = red[0] + bias[0];
}

// ---------------- host orchestration ----------------
// d_in layout (jax tree flatten, sorted dict keys):
// 0 pcd, 1 qry, 2 anc,
// 3 fc1.be, 4 fc1.conv.b, 5 fc1.conv.w[128,132], 6 fc1.g,
// 7 fc2.be, 8 fc2.conv.b, 9 fc2.conv.w[128,128], 10 fc2.g,
// 11 fc_out.b[1], 12 fc_out.w[6144],
// 13 + l*12: in_b, in_w[384,128], l1_b, l1_w[2048,128], l2_b, l2_w[128,2048],
//            ln1_b, ln1_g, ln2_b, ln2_g, out_b, out_w[128,128]        (l = 0..5)
// 85 + i*5: fc0_b, fc0_w[128,256], fc1_b, fc1_w[128,128], sc_w[128,256] (i = 0..4)
// 110 fc_c.b, 111 fc_c.w[128,128], 112 fc_pos.b[256], 113 fc_pos.w[256,4]
extern "C" void kernel_launch(void* const* d_in, const int* in_sizes, int n_in,
                              void* d_out, int out_size, void* d_ws, size_t ws_size,
                              hipStream_t stream)
{
  (void)in_sizes; (void)n_in; (void)out_size; (void)ws_size;
  const float* pcd = (const float*)d_in[0];
  const float* qry = (const float*)d_in[1];
  const float* anc = (const float*)d_in[2];
  auto F = [&](int i) { return (const float*)d_in[i]; };

  char* ws = (char*)d_ws;
  size_t off = 0;
  auto alloc = [&](size_t bytes) -> void* {
    off = (off + 255) & ~(size_t)255;
    void* p = ws + off;
    off += bytes;
    return p;
  };
  auto CVT = [&](int idx, size_t n) -> __bf16* {
    __bf16* dst = (__bf16*)alloc(n * sizeof(__bf16));
    k_cvt_bf16<<<(int)((n + 255) / 256), 256, 0, stream>>>(F(idx), dst, (long)n);
    return dst;
  };
  auto GEMM = [&](const __bf16* A, const __bf16* W, const float* bias,
                  const float* Cin, float* C, __bf16* Obf,
                  int M, int K, int N, int reluIn) {
    int blocks = (M / GBM) * (N / GBN);
    if (reluIn)
      k_gemm_bf16<true><<<blocks, 256, 0, stream>>>(A, W, bias, Cin, C, Obf, M, K, N);
    else
      k_gemm_bf16<false><<<blocks, 256, 0, stream>>>(A, W, bias, Cin, C, Obf, M, K, N);
  };

  // ---- bf16 weights ----
  __bf16 *wb_fc0[5], *wb_fc1[5], *wb_sc[5];
  for (int i = 0; i < 5; ++i) {
    int base = 85 + i * 5;
    wb_fc0[i] = CVT(base + 1, 128 * 256);
    wb_fc1[i] = CVT(base + 3, 128 * 128);
    wb_sc[i]  = CVT(base + 4, 128 * 256);
  }
  __bf16* wb_fcc = CVT(111, 128 * 128);
  __bf16* wb_c1 = (__bf16*)alloc((size_t)128 * KCAT * sizeof(__bf16));
  k_cvt_pad<<<(128 * KCAT + 255) / 256, 256, 0, stream>>>(F(5), wb_c1, 128, 132, KCAT);
  __bf16* wb_c2 = CVT(9, 128 * 128);
  __bf16 *wb_in[NL], *wb_op[NL], *wb_l1[NL], *wb_l2[NL];
  for (int l = 0; l < NL; ++l) {
    int base = 13 + l * 12;
    wb_in[l] = CVT(base + 1,  (size_t)3 * DM * DM);
    wb_l1[l] = CVT(base + 3,  (size_t)DFF * DM);
    wb_l2[l] = CVT(base + 5,  (size_t)DM * DFF);
    wb_op[l] = CVT(base + 11, (size_t)DM * DM);
  }

  // ---- activations ----
  float*  feat_qh = (float*)alloc((size_t)R_TOT * 4 * sizeof(float));
  float*  feat_aq = (float*)alloc((size_t)SAMPLES * 4 * sizeof(float));
  __bf16* xbf  = (__bf16*)alloc((size_t)RCHUNK * 256 * sizeof(__bf16));
  __bf16* h1b  = (__bf16*)alloc((size_t)RCHUNK * HID * sizeof(__bf16));
  float*  of   = (float*) alloc((size_t)RCHUNK * HID * sizeof(float));
  __bf16* pool = (__bf16*)alloc((size_t)SAMPLES * HID * sizeof(__bf16));
  float*  fl   = (float*) alloc((size_t)SAMPLES * HID * sizeof(float));
  __bf16* catb = (__bf16*)alloc((size_t)SAMPLES * KCAT * sizeof(__bf16));
  float*  y1   = (float*) alloc((size_t)SAMPLES * (3 * DM) * sizeof(float)); // qkv / conv scratch
  __bf16* hbf  = (__bf16*)alloc((size_t)SAMPLES * HID * sizeof(__bf16));
  float*  xf   = (float*) alloc((size_t)SAMPLES * DM * sizeof(float));
  __bf16* xbT  = (__bf16*)alloc((size_t)SAMPLES * DM * sizeof(__bf16));
  float*  x1f  = (float*) alloc((size_t)SAMPLES * DM * sizeof(float));
  __bf16* atb  = (__bf16*)alloc((size_t)SAMPLES * DM * sizeof(__bf16));
  float*  prj  = (float*) alloc((size_t)SAMPLES * DM * sizeof(float));
  __bf16* ffb  = (__bf16*)alloc((size_t)SAMPLES * DFF * sizeof(__bf16));

  // ---- 1) cone cast + top-32 + relative features ----
  k_cone_topk<<<SAMPLES, 256, 0, stream>>>(pcd, qry, anc, feat_qh, feat_aq);

  // ---- 2) ResnetPointnet (chunked over samples) ----
  for (int ch = 0; ch < NCHUNK; ++ch) {
    const size_t SB = (size_t)ch * SCHUNK;
    const size_t RB = SB * N_LOC;
    k_fc_pos<<<RCHUNK, 256, 0, stream>>>(feat_qh + RB * 4, F(113), F(112), xbf, RCHUNK);
    for (int blk = 0; blk < 5; ++blk) {
      int base = 85 + blk * 5;
      // h1 = relu(fc0(relu(x)))  (fused relu-out -> bf16)
      GEMM(xbf, wb_fc0[blk], F(base + 0), nullptr, nullptr, h1b, RCHUNK, 256, HID, 1);
      // o = fc1(h1) + shortcut(x)
      GEMM(h1b, wb_fc1[blk], F(base + 2), nullptr, of, nullptr, RCHUNK, HID, HID, 0);
      GEMM(xbf, wb_sc[blk], nullptr, of, of, nullptr, RCHUNK, 256, HID, 0);
      if (blk < 4) k_pool_concat<<<SCHUNK, HID, 0, stream>>>(of, xbf);
    }
    k_pool_relu<<<SCHUNK, HID, 0, stream>>>(of, pool + SB * HID);
  }
  GEMM(pool, wb_fcc, F(110), nullptr, fl, nullptr, SAMPLES, HID, HID, 0);

  // ---- 3) concat + conv-bn-relu x2 (rows in (b,a,q) order) ----
  k_concat_fc1<<<SAMPLES, KCAT, 0, stream>>>(feat_aq, fl, catb);
  GEMM(catb, wb_c1, F(4), nullptr, y1, nullptr, SAMPLES, KCAT, HID, 0);
  k_bn_relu<<<(SAMPLES * HID + 255) / 256, 256, 0, stream>>>(y1, F(6), F(3), hbf, (long)SAMPLES * HID);
  GEMM(hbf, wb_c2, F(8), nullptr, y1, nullptr, SAMPLES, HID, HID, 0);
  k_bn_relu_tr<<<SAMPLES, HID, 0, stream>>>(y1, F(10), F(7), xf, xbT);  // -> (b,q,a) rows

  // ---- 4) 6x transformer encoder layers ----
  for (int l = 0; l < NL; ++l) {
    int base = 13 + l * 12;
    GEMM(xbT, wb_in[l], F(base + 0), nullptr, y1, nullptr, SAMPLES, DM, 3 * DM, 0);  // qkv
    k_attn<<<N_BS * N_QRY * NH, 64, 0, stream>>>(y1, atb);
    GEMM(atb, wb_op[l], F(base + 10), nullptr, prj, nullptr, SAMPLES, DM, DM, 0);    // out_proj
    k_add_ln<<<SAMPLES, DM, 0, stream>>>(xf, prj, F(base + 7), F(base + 6), x1f, xbT);
    // ff: relu(lin1(x)) fused -> bf16, then lin2
    GEMM(xbT, wb_l1[l], F(base + 2), nullptr, nullptr, ffb, SAMPLES, DM, DFF, 0);
    GEMM(ffb, wb_l2[l], F(base + 4), nullptr, prj, nullptr, SAMPLES, DFF, DM, 0);
    k_add_ln<<<SAMPLES, DM, 0, stream>>>(x1f, prj, F(base + 9), F(base + 8), xf, xbT);
  }

  // ---- 5) output head ----
  k_out<<<N_BS * N_QRY, 256, 0, stream>>>(xf, F(12), F(11), (float*)d_out);
}
